// OptimizedDifferentiableDenseHGTConv_39865886442164
// MI455X (gfx1250) — matrix operand order repaired and checked by reading, not compiled, not run on hardware
//
#include <hip/hip_runtime.h>
#include <hip/hip_bf16.h>
#include <math.h>

typedef __attribute__((ext_vector_type(16))) __bf16 v16bf;
typedef __attribute__((ext_vector_type(8)))  float  v8f;

#define BB   4
#define NN   384
#define DIN  128
#define DOUTC 128
#define TT   4
#define RR   4
#define HH   8
#define DKK  16
#define INV_SQRT_DK 0.25f   // DK=16 -> sqrt=4

__device__ __forceinline__ v8f wmma_bf16(v16bf a, v16bf b, v8f c) {
  // v_wmma_f32_16x16x32_bf16 : D = A(16x32) * B(32x16) + C
  return __builtin_amdgcn_wmma_f32_16x16x32_bf16(false, a, false, b, (short)0, c, false, false);
}

// ---------------------------------------------------------------------------
// Kernel 1: fused Q/K/V projections (fp32 GEMV per output element).
// q is written directly as bf16 "Qc" in [B,H,N, (t*16+dk)=64] layout for WMMA A.
// k,v stay fp32 (feed relation projections).
// ---------------------------------------------------------------------------
__global__ void k_proj(const float* __restrict__ x,
                       const float* __restrict__ qw, const float* __restrict__ qb,
                       const float* __restrict__ kw, const float* __restrict__ kb,
                       const float* __restrict__ vw, const float* __restrict__ vb,
                       __bf16* __restrict__ Qc, float* __restrict__ kk, float* __restrict__ vv) {
  int idx = blockIdx.x * blockDim.x + threadIdx.x;
  const int per = BB*NN*TT*DOUTC;           // 786432
  int which = idx / per;
  int a = idx - which * per;                // a = ((b*N+n)*T + t)*128 + d
  int d  = a & 127;
  int t  = (a >> 7) & 3;
  int bn = a >> 9;
  const float* w; const float* bias;
  if      (which == 0) { w = qw; bias = qb; }
  else if (which == 1) { w = kw; bias = kb; }
  else                 { w = vw; bias = vb; }
  const float* xp = x + (size_t)bn * DIN;
  const float* wp = w + ((size_t)t * DIN) * DOUTC + d;
  float acc = bias[t*DOUTC + d];
  #pragma unroll 8
  for (int i = 0; i < DIN; ++i) acc += xp[i] * wp[(size_t)i * DOUTC];
  if (which == 0) {
    int b = bn / NN, n = bn - b*NN;
    int h = d >> 4, dk = d & 15;
    Qc[(((size_t)(b*HH + h))*NN + n)*64 + t*16 + dk] = (__bf16)acc;
  } else if (which == 1) kk[a] = acc;
  else                   vv[a] = acc;
}

// ---------------------------------------------------------------------------
// Kernel 2: relation projections.
// Kc[r,s,b,h,j, t*16+dk] = nt[b,j,t] * (k[b,j,s,h,:] @ rel_att[r,h])  (bf16)
// VcT[r,s,b,h,dk, j]     = (v[b,j,s,h,:] @ rel_msg[r,h])              (bf16)
// ---------------------------------------------------------------------------
__global__ void k_relproj(const float* __restrict__ nt,
                          const float* __restrict__ kk, const float* __restrict__ vv,
                          const float* __restrict__ rel_att, const float* __restrict__ rel_msg,
                          __bf16* __restrict__ Kc, __bf16* __restrict__ VcT) {
  int idx = blockIdx.x * blockDim.x + threadIdx.x;
  // idx = ((((r*B + b)*N + j)*S + s)*H + h)*16 + dk
  int dk = idx & 15;
  int h  = (idx >> 4) & 7;
  int s  = (idx >> 7) & 3;
  int rbj = idx >> 9;
  int j  = rbj % NN;
  int rb = rbj / NN;
  int b  = rb & 3;
  int r  = rb >> 2;
  const float* kp = kk + (((size_t)(b*NN + j))*TT + s)*DOUTC + h*16;
  const float* vp = vv + (((size_t)(b*NN + j))*TT + s)*DOUTC + h*16;
  const float* ra = rel_att + ((size_t)(r*HH + h))*256 + dk;  // [h,e,dk]
  const float* rm = rel_msg + ((size_t)(r*HH + h))*256 + dk;
  float kr = 0.f, vr = 0.f;
  #pragma unroll
  for (int e = 0; e < 16; ++e) { kr += kp[e]*ra[e*16]; vr += vp[e]*rm[e*16]; }
  int rs = r*4 + s;
  size_t kbase = ((((size_t)rs*BB + b)*HH + h)*NN + j)*64 + (size_t)dk;
  const float* ntp = nt + (size_t)(b*NN + j)*TT;
  #pragma unroll
  for (int t = 0; t < 4; ++t) Kc[kbase + t*16] = (__bf16)(ntp[t] * kr);
  VcT[((((size_t)rs*BB + b)*HH + h)*16 + dk)*NN + j] = (__bf16)vr;
}

// ---------------------------------------------------------------------------
// Pass 1: att_sum[b,h,j] = sum_i res_att[b,i,j,h]
// block = (b,h, j-tile of 32), single wave32; loop over 24 i-tiles.
// ---------------------------------------------------------------------------
__global__ void k_pass1(const float* __restrict__ nt, const float* __restrict__ adj,
                        const float* __restrict__ edge, const __bf16* __restrict__ Qc,
                        const __bf16* __restrict__ Kc, const float* __restrict__ rel_pri,
                        float* __restrict__ att_sum) {
  int blk = blockIdx.x;
  int jt = (blk % (NN/32)) * 32;
  int bh = blk / (NN/32);
  int h = bh & 7, b = bh >> 3;
  int lane = threadIdx.x;
  int lm = lane & 15, hi = lane >> 4;
  __shared__ float Glds[16][16][32];
  __shared__ float masklds[4][16][32];
  __shared__ float ntil[16][4];
  float rpn[4];
  #pragma unroll
  for (int r = 0; r < 4; ++r) rpn[r] = rel_pri[r*HH + h] * INV_SQRT_DK;
  float colacc = 0.f;
  int kb = hi*8;
  for (int it = 0; it < NN/16; ++it) {
    int ibase = it*16;
    __syncthreads();
    for (int cell = lane; cell < 512; cell += 32) {
      int m = cell >> 5, jc = cell & 31;
      size_t ij = ((size_t)b*NN + ibase + m)*NN + jt + jc;
      float av = adj[ij];
      const float* ep = edge + ij*RR;
      #pragma unroll
      for (int r = 0; r < 4; ++r) masklds[r][m][jc] = av * ep[r];
    }
    for (int u = lane; u < 64; u += 32) {
      int m = u >> 2, s = u & 3;
      ntil[m][s] = nt[((size_t)b*NN + ibase + m)*TT + s];
    }
    const __bf16* qp = Qc + (((size_t)(b*HH + h))*NN + ibase + lm)*64;
    v16bf a0, a1;
    #pragma unroll
    for (int e = 0; e < 8; ++e) {
      a0[e]   = qp[2*kb+e];      a0[8+e] = qp[2*kb+8+e];
      a1[e]   = qp[32+2*kb+e];   a1[8+e] = qp[40+2*kb+e];
    }
    __syncthreads();
    for (int rs = 0; rs < 16; ++rs) {
      #pragma unroll
      for (int sub = 0; sub < 2; ++sub) {
        const __bf16* kp = Kc + ((((size_t)rs*BB + b)*HH + h)*NN + jt + sub*16 + lm)*64 + hi*16;
        v16bf b0, b1;
        #pragma unroll
        for (int e = 0; e < 16; ++e) { b0[e] = kp[e]; b1[e] = kp[32+e]; }
        v8f acc = {};
        acc = wmma_bf16(a0, b0, acc);
        acc = wmma_bf16(a1, b1, acc);
        #pragma unroll
        for (int v = 0; v < 8; ++v) Glds[rs][v + 8*hi][sub*16 + lm] = acc[v];
      }
    }
    __syncthreads();
    // S-combine; lane owns column jc = lane
    for (int m = 0; m < 16; ++m) {
      float sacc = 0.f;
      #pragma unroll
      for (int rs = 0; rs < 16; ++rs)
        sacc += masklds[rs>>2][m][lane] * rpn[rs>>2] * ntil[m][rs&3] * Glds[rs][m][lane];
      colacc += sacc;
    }
  }
  att_sum[((size_t)b*HH + h)*NN + jt + lane] = colacc;
}

// ---------------------------------------------------------------------------
// Pass 2: recompute G tiles, normalize (w = S/att_sum, guarded), and
// agg[b,i,h,:] += sum_j w*mask*rp'*nt_i_s*G * VcT  via a second WMMA chain.
// block = (b,h, i-tile of 16), single wave32; loop over 12 j-tiles of 32.
// ---------------------------------------------------------------------------
__global__ void k_pass2(const float* __restrict__ nt, const float* __restrict__ adj,
                        const float* __restrict__ edge, const __bf16* __restrict__ Qc,
                        const __bf16* __restrict__ Kc, const __bf16* __restrict__ VcT,
                        const float* __restrict__ rel_pri, const float* __restrict__ att_sum,
                        float* __restrict__ agg) {
  int blk = blockIdx.x;
  int it = blk % (NN/16);
  int bh = blk / (NN/16);
  int h = bh & 7, b = bh >> 3;
  int ibase = it*16;
  int lane = threadIdx.x;
  int lm = lane & 15, hi = lane >> 4;
  __shared__ float Glds[16][16][32];
  __shared__ float masklds[4][16][32];
  __shared__ float wlds[16][32];
  __shared__ float ntil[16][4];
  float rpn[4];
  #pragma unroll
  for (int r = 0; r < 4; ++r) rpn[r] = rel_pri[r*HH + h] * INV_SQRT_DK;
  for (int u = lane; u < 64; u += 32) {
    int m = u >> 2, s = u & 3;
    ntil[m][s] = nt[((size_t)b*NN + ibase + m)*TT + s];
  }
  int kb = hi*8;
  const __bf16* qp = Qc + (((size_t)(b*HH + h))*NN + ibase + lm)*64;
  v16bf a0, a1;
  #pragma unroll
  for (int e = 0; e < 8; ++e) {
    a0[e]   = qp[2*kb+e];      a0[8+e] = qp[2*kb+8+e];
    a1[e]   = qp[32+2*kb+e];   a1[8+e] = qp[40+2*kb+e];
  }
  v8f aggacc = {};
  for (int jt = 0; jt < NN; jt += 32) {
    __syncthreads();
    for (int cell = lane; cell < 512; cell += 32) {
      int m = cell >> 5, jc = cell & 31;
      size_t ij = ((size_t)b*NN + ibase + m)*NN + jt + jc;
      float av = adj[ij];
      const float* ep = edge + ij*RR;
      #pragma unroll
      for (int r = 0; r < 4; ++r) masklds[r][m][jc] = av * ep[r];
    }
    float attc = att_sum[((size_t)b*HH + h)*NN + jt + lane];
    __syncthreads();
    // Phase A: all 16 G tiles into LDS
    for (int rs = 0; rs < 16; ++rs) {
      #pragma unroll
      for (int sub = 0; sub < 2; ++sub) {
        const __bf16* kp = Kc + ((((size_t)rs*BB + b)*HH + h)*NN + jt + sub*16 + lm)*64 + hi*16;
        v16bf b0, b1;
        #pragma unroll
        for (int e = 0; e < 16; ++e) { b0[e] = kp[e]; b1[e] = kp[32+e]; }
        v8f acc = {};
        acc = wmma_bf16(a0, b0, acc);
        acc = wmma_bf16(a1, b1, acc);
        #pragma unroll
        for (int v = 0; v < 8; ++v) Glds[rs][v + 8*hi][sub*16 + lm] = acc[v];
      }
    }
    __syncthreads();
    // Phase B: total score S -> normalized weight w (column jc = lane)
    for (int m = 0; m < 16; ++m) {
      float sacc = 0.f;
      #pragma unroll
      for (int rs = 0; rs < 16; ++rs)
        sacc += masklds[rs>>2][m][lane] * rpn[rs>>2] * ntil[m][rs&3] * Glds[rs][m][lane];
      wlds[m][lane] = (attc > 1e-6f) ? (sacc / attc) : 0.f;
    }
    __syncthreads();
    // Phase C: agg += C_{r,s}(16x32) x VcT(32x16) via WMMA
    for (int rs = 0; rs < 16; ++rs) {
      int r = rs >> 2, s = rs & 3;
      v16bf ca;
      #pragma unroll
      for (int e = 0; e < 8; ++e) {
        int j0 = 2*kb + e;
        int j1 = 2*kb + 8 + e;
        ca[e]   = (__bf16)(wlds[lm][j0]*masklds[r][lm][j0]*rpn[r]*ntil[lm][s]*Glds[rs][lm][j0]);
        ca[8+e] = (__bf16)(wlds[lm][j1]*masklds[r][lm][j1]*rpn[r]*ntil[lm][s]*Glds[rs][lm][j1]);
      }
      const __bf16* vp = VcT + ((((size_t)rs*BB + b)*HH + h)*16 + lm)*NN + jt + hi*16;
      v16bf vb;
      #pragma unroll
      for (int e = 0; e < 16; ++e) vb[e] = vp[e];
      aggacc = wmma_bf16(ca, vb, aggacc);
    }
  }
  #pragma unroll
  for (int v = 0; v < 8; ++v)
    agg[(((size_t)b*NN + ibase + v + 8*hi)*HH + h)*16 + lm] = aggacc[v];
}

// ---------------------------------------------------------------------------
// Epilogue: per (b,n): trans = gelu(agg @ aw_t + ab_t); skip-mix; layernorm;
// type-weighted sum. 128 threads = one output channel each.
// ---------------------------------------------------------------------------
__global__ void k_epilogue(const float* __restrict__ agg, const float* __restrict__ x,
                           const float* __restrict__ nt,  const float* __restrict__ aw,
                           const float* __restrict__ ab,  const float* __restrict__ normw,
                           const float* __restrict__ normb,const float* __restrict__ skip,
                           float* __restrict__ out) {
  int bn = blockIdx.x;
  int o  = threadIdx.x;
  __shared__ float aggs[128];
  __shared__ float red[128];
  aggs[o] = agg[(size_t)bn*DOUTC + o];
  __syncthreads();
  float xv = x[(size_t)bn*DOUTC + o];
  float outacc = 0.f;
  for (int t = 0; t < TT; ++t) {
    float z = ab[t*DOUTC + o];
    const float* awp = aw + (size_t)t*DOUTC*DOUTC + o;
    #pragma unroll 8
    for (int d = 0; d < DOUTC; ++d) z += aggs[d] * awp[(size_t)d*DOUTC];
    float g = 0.5f * z * (1.f + erff(z * 0.70710678118654752f));  // exact gelu
    float alpha = 1.f / (1.f + expf(-skip[t]));
    float res = g * alpha + xv * (1.f - alpha);
    // mean
    red[o] = res; __syncthreads();
    for (int off = 64; off > 0; off >>= 1) { if (o < off) red[o] += red[o+off]; __syncthreads(); }
    float mean = red[0] * (1.f/128.f); __syncthreads();
    // var (population)
    red[o] = res*res; __syncthreads();
    for (int off = 64; off > 0; off >>= 1) { if (o < off) red[o] += red[o+off]; __syncthreads(); }
    float var = red[0] * (1.f/128.f) - mean*mean; __syncthreads();
    float nrm = (res - mean) * rsqrtf(var + 1e-5f) * normw[t*DOUTC + o] + normb[t*DOUTC + o];
    outacc += nt[(size_t)bn*TT + t] * nrm;
  }
  out[(size_t)bn*DOUTC + o] = outacc;
}

// ---------------------------------------------------------------------------
extern "C" void kernel_launch(void* const* d_in, const int* in_sizes, int n_in,
                              void* d_out, int out_size, void* d_ws, size_t ws_size,
                              hipStream_t stream) {
  (void)in_sizes; (void)n_in; (void)out_size; (void)ws_size;
  const float* x     = (const float*)d_in[0];
  const float* nt    = (const float*)d_in[1];
  const float* adj   = (const float*)d_in[2];
  const float* edge  = (const float*)d_in[3];
  const float* kw    = (const float*)d_in[4];
  const float* kb_   = (const float*)d_in[5];
  const float* qw    = (const float*)d_in[6];
  const float* qb_   = (const float*)d_in[7];
  const float* vw    = (const float*)d_in[8];
  const float* vb_   = (const float*)d_in[9];
  const float* aw    = (const float*)d_in[10];
  const float* ab_   = (const float*)d_in[11];
  const float* normw = (const float*)d_in[12];
  const float* normb = (const float*)d_in[13];
  const float* rel_pri = (const float*)d_in[14];
  const float* rel_att = (const float*)d_in[15];
  const float* rel_msg = (const float*)d_in[16];
  const float* skip  = (const float*)d_in[17];
  float* out = (float*)d_out;

  char* ws = (char*)d_ws;
  __bf16* Qc  = (__bf16*)ws;  ws += (size_t)BB*HH*NN*64*2;            // 1.5 MB
  float*  kk  = (float*)ws;   ws += (size_t)BB*NN*TT*DOUTC*4;         // 3 MB
  float*  vv  = (float*)ws;   ws += (size_t)BB*NN*TT*DOUTC*4;         // 3 MB
  __bf16* Kc  = (__bf16*)ws;  ws += (size_t)RR*TT*BB*HH*NN*64*2;      // 24 MB
  __bf16* VcT = (__bf16*)ws;  ws += (size_t)RR*TT*BB*HH*16*NN*2;      // 6 MB
  float*  att = (float*)ws;   ws += (size_t)BB*HH*NN*4;               // 48 KB
  float*  agg = (float*)ws;   ws += (size_t)BB*NN*DOUTC*4;            // 768 KB

  k_proj<<<(3*BB*NN*TT*DOUTC)/256, 256, 0, stream>>>(x, qw, qb_, kw, kb_, vw, vb_, Qc, kk, vv);
  k_relproj<<<(RR*BB*NN*TT*HH*16)/256, 256, 0, stream>>>(nt, kk, vv, rel_att, rel_msg, Kc, VcT);
  k_pass1<<<BB*HH*(NN/32), 32, 0, stream>>>(nt, adj, edge, Qc, Kc, rel_pri, att);
  k_pass2<<<BB*HH*(NN/16), 32, 0, stream>>>(nt, adj, edge, Qc, Kc, VcT, rel_pri, att, agg);
  k_epilogue<<<BB*NN, 128, 0, stream>>>(agg, x, nt, aw, ab_, normw, normb, skip, out);
}
